// SparseMultiHeadAttention_27779848471222
// MI455X (gfx1250) — matrix operand order repaired and checked by reading, not compiled, not run on hardware
//
#include <hip/hip_runtime.h>

typedef __attribute__((ext_vector_type(16))) _Float16 v16h;
typedef __attribute__((ext_vector_type(8)))  _Float16 v8h;
typedef __attribute__((ext_vector_type(8)))  float    v8f;

#define WMMA_F16(a, b, c) \
  __builtin_amdgcn_wmma_f32_16x16x32_f16(false, (a), false, (b), (short)0, (c), false, false)

// ---------------------------------------------------------------------------
// Fragment loaders matching the CDNA5 ISA VGPR layouts (CDNA5 ISA).
// A (16x32 f16): lane L holds row M=L%16, K runs [ks..ks+7] and [ks+16..ks+23]
// with ks = 8*(L/16)  -> two contiguous v8h loads per lane.
// B (32x16 f16): lane L holds col N=L%16, K runs [ks..ks+7] and [ks+16..ks+23]
// with ks = 8*(L/16)  -> two contiguous v8h loads per lane.
// C/D (16x16 f32): VGPR r, lane L -> (M = 8*(L/16)+r, N = L%16).
// ---------------------------------------------------------------------------
static __device__ __forceinline__ v16h load_a_frag(const _Float16* __restrict__ base,
                                                   int ld, int row0, int kb) {
  const int lane = threadIdx.x & 31;
  const int m  = row0 + (lane & 15);
  const int ks = (lane >> 4) << 3;
  const _Float16* p = base + (long)m * ld + kb + ks;
  v16h a;
  *(v8h*)&a       = *(const v8h*)p;
  *((v8h*)&a + 1) = *(const v8h*)(p + 16);
  return a;
}

static __device__ __forceinline__ v16h load_b_frag(const _Float16* __restrict__ w,
                                                   int ldw, int n0, int kb) {
  const int lane = threadIdx.x & 31;
  const int n  = n0 + (lane & 15);
  const int ks = (lane >> 4) << 3;
  const _Float16* p = w + (long)n * ldw + kb + ks;
  v16h b;
  *(v8h*)&b       = *(const v8h*)p;
  *((v8h*)&b + 1) = *(const v8h*)(p + 16);
  return b;
}

// ---------------------------------------------------------------------------
// f32 -> f16 conversion (x and the four weight matrices)
// ---------------------------------------------------------------------------
__global__ void cvt_f32_to_f16_kernel(const float* __restrict__ in,
                                      _Float16* __restrict__ out, int n) {
  int i = (blockIdx.x * blockDim.x + threadIdx.x) * 4;
  if (i + 3 < n) {
    float4 v = *(const float4*)(in + i);
    out[i + 0] = (_Float16)v.x;
    out[i + 1] = (_Float16)v.y;
    out[i + 2] = (_Float16)v.z;
    out[i + 3] = (_Float16)v.w;
  } else {
    for (int j = i; j < n; ++j) out[j] = (_Float16)in[j];
  }
}

// ---------------------------------------------------------------------------
// Generic WMMA GEMM:  out = A(MxK,f16) * W(NxK,f16)^T + bias
//  - out16 != null, transpose_S == 0 : f16 row-major [M x N]
//  - out16 != null, transpose_S == S : f16 "Vt" store out[(b*N + n)*S + s]
//  - out32 != null                   : f32 row-major [M x N]
// Block: 256 threads = 8 waves arranged 2(M) x 4(N); wave tile 32x64;
// workgroup tile 64 x 256.  K-loop step 32 -> 8 wmma per iteration per wave.
// ---------------------------------------------------------------------------
__global__ void gemm_wmma_kernel(const _Float16* __restrict__ A,
                                 const _Float16* __restrict__ W,
                                 const float* __restrict__ bias,
                                 _Float16* __restrict__ out16,
                                 float* __restrict__ out32,
                                 int M, int N, int K, int transpose_S) {
  const int lane = threadIdx.x & 31;
  const int wid  = threadIdx.x >> 5;
  const int mw = wid & 1, nw = wid >> 1;
  const int row0 = blockIdx.x * 64  + mw * 32;
  const int col0 = blockIdx.y * 256 + nw * 64;

  v8f zero = {};
  v8f c[2][4];
#pragma unroll
  for (int i = 0; i < 2; ++i)
#pragma unroll
    for (int j = 0; j < 4; ++j) c[i][j] = zero;

  for (int kb = 0; kb < K; kb += 32) {
    if (kb + 32 < K) {
      __builtin_prefetch(A + (long)(row0 + (lane & 15)) * K + kb + 32, 0, 1);
      __builtin_prefetch(W + (long)(col0 + (lane & 15)) * K + kb + 32, 0, 1);
    }
    v16h a0 = load_a_frag(A, K, row0, kb);
    v16h a1 = load_a_frag(A, K, row0 + 16, kb);
#pragma unroll
    for (int j = 0; j < 4; ++j) {
      v16h bf = load_b_frag(W, K, col0 + j * 16, kb);
      c[0][j] = WMMA_F16(a0, bf, c[0][j]);
      c[1][j] = WMMA_F16(a1, bf, c[1][j]);
    }
  }

  const int n_in = lane & 15;
  const int mb8  = (lane >> 4) << 3;
#pragma unroll
  for (int i = 0; i < 2; ++i) {
#pragma unroll
    for (int j = 0; j < 4; ++j) {
      const int n = col0 + j * 16 + n_in;
      const float bv = bias ? bias[n] : 0.f;
#pragma unroll
      for (int r = 0; r < 8; ++r) {
        const int m = row0 + i * 16 + mb8 + r;
        const float val = c[i][j][r] + bv;
        if (out32) {
          out32[(long)m * N + n] = val;
        } else if (transpose_S) {
          const int bb = m / transpose_S, ss = m % transpose_S;
          out16[((long)bb * N + n) * transpose_S + ss] = (_Float16)val;
        } else {
          out16[(long)m * N + n] = (_Float16)val;
        }
      }
    }
  }
}

// ---------------------------------------------------------------------------
// Fused sparse attention: one workgroup owns a 16-query strip of one batch,
// loops over all 16 heads.  LDS (dynamic, 266240 B total, fits 320KB WGP LDS):
//   sc  : 16 x 2048 f32  (128 KB)  scores -> probs, in place
//   avg : 16 x 2048 f32  (128 KB)  running mean over heads
//   red : 4 x 16 x 16 f32 (4 KB)   cross-wave partial reduction for attn@V
// ---------------------------------------------------------------------------
__global__ void attn_kernel(const _Float16* __restrict__ Q,
                            const _Float16* __restrict__ Km,
                            const _Float16* __restrict__ Vt,
                            const float* __restrict__ temperature,
                            _Float16* __restrict__ ctx,
                            float* __restrict__ avg_out) {
  constexpr int S = 2048, D = 1024, H = 16, HD = 64, TOPK = 409;
  extern __shared__ float smem[];
  float* sc  = smem;
  float* avg = smem + 16 * S;
  float* red = avg + 16 * S;

  const int tid  = threadIdx.x;
  const int lane = tid & 31;
  const int wid  = tid >> 5;
  const int b  = blockIdx.x >> 7;           // / (S/16)
  const int q0 = (blockIdx.x & 127) << 4;   // * 16

  const int n_in = lane & 15;
  const int mb8  = (lane >> 4) << 3;

  for (int i = tid; i < 16 * S; i += 256) avg[i] = 0.f;
  __syncthreads();

  for (int h = 0; h < H; ++h) {
    float t = temperature[h];
    t = t < 0.1f ? 0.1f : t;
    const float sf = 0.125f / t;  // hd^-0.5 / temp
    const int hc = h * HD;

    // ---- scores = (q . k^T) * sf ; 8 waves split the 128 key blocks ----
    v16h aq0 = load_a_frag(Q, D, b * S + q0, hc);
    v16h aq1 = load_a_frag(Q, D, b * S + q0, hc + 32);
    for (int kb = wid * 16; kb < S; kb += 128) {
      v16h bk0 = load_b_frag(Km, D, b * S + kb, hc);
      v16h bk1 = load_b_frag(Km, D, b * S + kb, hc + 32);
      v8f cf = {};
      cf = WMMA_F16(aq0, bk0, cf);
      cf = WMMA_F16(aq1, bk1, cf);
#pragma unroll
      for (int r = 0; r < 8; ++r)
        sc[(mb8 + r) * S + kb + n_in] = cf[r] * sf;
    }
    __syncthreads();

    // ---- exact top-k threshold (radix select) + masked softmax, per row ----
    for (int rrow = wid; rrow < 16; rrow += 8) {
      float* row = sc + rrow * S;
      float mx = -3.4e38f;
      for (int i = lane; i < S; i += 32) mx = fmaxf(mx, row[i]);
#pragma unroll
      for (int off = 16; off; off >>= 1) mx = fmaxf(mx, __shfl_xor(mx, off, 32));

      unsigned T = 0u;
#pragma unroll 1
      for (int bit = 31; bit >= 0; --bit) {
        const unsigned cand = T | (1u << bit);
        int cnt = 0;
        for (int i = lane; i < S; i += 32) {
          unsigned u = __float_as_uint(row[i]);
          u ^= (unsigned)((int)u >> 31) | 0x80000000u;  // monotonic map
          cnt += (u >= cand) ? 1 : 0;
        }
#pragma unroll
        for (int off = 16; off; off >>= 1) cnt += __shfl_xor(cnt, off, 32);
        if (cnt >= TOPK) T = cand;
      }

      float sum = 0.f;
      for (int i = lane; i < S; i += 32) {
        const float x = row[i];
        unsigned u = __float_as_uint(x);
        u ^= (unsigned)((int)u >> 31) | 0x80000000u;
        const float e = (u >= T) ? __expf(x - mx) : 0.f;
        row[i] = e;
        sum += e;
      }
#pragma unroll
      for (int off = 16; off; off >>= 1) sum += __shfl_xor(sum, off, 32);
      const float inv = 1.f / sum;
      for (int i = lane; i < S; i += 32) {
        const float p = row[i] * inv;
        row[i] = p;
        avg[rrow * S + i] += p * (1.f / 16.f);
      }
    }
    __syncthreads();

    // ---- head_out = probs @ V : waves split (ntile in 0..3) x (K half) ----
    const int ntile = wid & 3;
    const int khalf = wid >> 2;
    const int ks8 = mb8;  // same 8*(lane/16)
    v8f co = {};
    const int kbase0 = khalf * (S / 2);
    for (int kc = 0; kc < S / 2; kc += 32) {
      const int keyb = kbase0 + kc;
      const float* pr = sc + (lane & 15) * S + keyb + ks8;
      v16h pa;
#pragma unroll
      for (int e = 0; e < 8; ++e) pa[e] = (_Float16)pr[e];
#pragma unroll
      for (int e = 0; e < 8; ++e) pa[8 + e] = (_Float16)pr[16 + e];
      v16h bv = load_b_frag(Vt, S, b * D + hc + ntile * 16, keyb);
      co = WMMA_F16(pa, bv, co);
    }
    if (khalf) {
#pragma unroll
      for (int r = 0; r < 8; ++r)
        red[ntile * 256 + (mb8 + r) * 16 + n_in] = co[r];
    }
    __syncthreads();
    if (!khalf) {
#pragma unroll
      for (int r = 0; r < 8; ++r) {
        const float v = co[r] + red[ntile * 256 + (mb8 + r) * 16 + n_in];
        ctx[(long)(b * S + q0 + mb8 + r) * D + hc + ntile * 16 + n_in] = (_Float16)v;
      }
    }
    __syncthreads();
  }

  // ---- flush mean attention: 16 rows x S are contiguous in d_out ----
  float* dst = avg_out + (long)(b * S + q0) * S;
  for (int i = tid; i < 16 * S / 4; i += 256)
    ((float4*)dst)[i] = ((float4*)avg)[i];
}

// ---------------------------------------------------------------------------
extern "C" void kernel_launch(void* const* d_in, const int* in_sizes, int n_in,
                              void* d_out, int out_size, void* d_ws, size_t ws_size,
                              hipStream_t stream) {
  constexpr int B = 2, S = 2048, D = 1024;
  const int M = B * S;

  const float* x  = (const float*)d_in[0];
  const float* Wq = (const float*)d_in[1];
  const float* bq = (const float*)d_in[2];
  const float* Wk = (const float*)d_in[3];
  const float* bk = (const float*)d_in[4];
  const float* Wv = (const float*)d_in[5];
  const float* bv = (const float*)d_in[6];
  const float* Wo = (const float*)d_in[7];
  const float* bo = (const float*)d_in[8];
  const float* temperature = (const float*)d_in[9];

  char* ws = (char*)d_ws;
  size_t off = 0;
  auto alloc = [&](size_t bytes) {
    char* p = ws + off;
    off += (bytes + 255) & ~(size_t)255;
    return p;
  };
  _Float16* x16   = (_Float16*)alloc((size_t)M * D * 2);
  _Float16* Wq16  = (_Float16*)alloc((size_t)D * D * 2);
  _Float16* Wk16  = (_Float16*)alloc((size_t)D * D * 2);
  _Float16* Wv16  = (_Float16*)alloc((size_t)D * D * 2);
  _Float16* Wo16  = (_Float16*)alloc((size_t)D * D * 2);
  _Float16* Q16   = (_Float16*)alloc((size_t)M * D * 2);
  _Float16* K16   = (_Float16*)alloc((size_t)M * D * 2);
  _Float16* Vt16  = (_Float16*)alloc((size_t)M * D * 2);  // [B*D, S]
  _Float16* ctx16 = (_Float16*)alloc((size_t)M * D * 2);

  // precision downconvert
  cvt_f32_to_f16_kernel<<<(M * D / 4 + 255) / 256, 256, 0, stream>>>(x, x16, M * D);
  cvt_f32_to_f16_kernel<<<(D * D / 4 + 255) / 256, 256, 0, stream>>>(Wq, Wq16, D * D);
  cvt_f32_to_f16_kernel<<<(D * D / 4 + 255) / 256, 256, 0, stream>>>(Wk, Wk16, D * D);
  cvt_f32_to_f16_kernel<<<(D * D / 4 + 255) / 256, 256, 0, stream>>>(Wv, Wv16, D * D);
  cvt_f32_to_f16_kernel<<<(D * D / 4 + 255) / 256, 256, 0, stream>>>(Wo, Wo16, D * D);

  dim3 gg(M / 64, D / 256);
  gemm_wmma_kernel<<<gg, 256, 0, stream>>>(x16, Wq16, bq, Q16, nullptr, M, D, D, 0);
  gemm_wmma_kernel<<<gg, 256, 0, stream>>>(x16, Wk16, bk, K16, nullptr, M, D, D, 0);
  gemm_wmma_kernel<<<gg, 256, 0, stream>>>(x16, Wv16, bv, Vt16, nullptr, M, D, D, S);

  float* outp = (float*)d_out;
  float* avgp = outp + (size_t)M * D;
  const size_t shmem = (size_t)(16 * S * 2 + 4 * 256) * sizeof(float);  // 266240 B
  attn_kernel<<<B * (S / 16), 256, shmem, stream>>>(Q16, K16, Vt16, temperature,
                                                    ctx16, avgp);

  gemm_wmma_kernel<<<gg, 256, 0, stream>>>(ctx16, Wo16, bo, nullptr, outp, M, D, D, 0);
}
